// DynamicMemoryPool_27058293965318
// MI455X (gfx1250) — compile-verified, matrix operand order checked by reading, not run on hardware
//
#include <hip/hip_runtime.h>
#include <hip/hip_bf16.h>

// ---------------- problem constants (from reference) ----------------
#define EDIM   768          // embedding dim
#define TOPK   8
#define NEGINF (-1.0e9f)

// ---------------- main-kernel tiling ----------------
#define QT     32           // queries per workgroup (2 WMMA row tiles)
#define NCH    64           // memory rows per chunk (4 waves x 16 cols)
#define ROWP   776          // padded LDS row length in bf16 (768 + 8), 16B-multiple
#define SIMS   65           // sim LDS row stride (floats), padded vs 64 banks
#define NTHR   128          // 4 waves (wave32)
#define KSTEPS (EDIM / 32)  // 24 WMMA k-steps

// LDS carve offsets (bytes)
#define QS_OFF    0
#define QS_BYTES  (QT * ROWP * 2)                 // 49,664
#define MS_OFF    (QS_OFF + QS_BYTES)
#define MS_BYTES  (NCH * ROWP * 2)                // 99,328
#define SIM_OFF   (MS_OFF + MS_BYTES)             // 148,992
#define SIM_BYTES (QT * SIMS * 4)                 // 8,320
#define MLAB_OFF  (SIM_OFF + SIM_BYTES)           // 157,312
#define MLAB_BYTES (NCH * 4)
#define SMEM_TOTAL (MLAB_OFF + MLAB_BYTES)        // 157,568 bytes
// post-loop aliases (over ms region, which is dead after the chunk loop)
#define MGV_OFF   MS_OFF
#define MGI_OFF   (MS_OFF + QT * 32 * 4)
#define WV_OFF    (MS_OFF + QT * 32 * 8)
#define WI_OFF    (WV_OFF + QT * TOPK * 4)

// CDNA5 async global->LDS path (gated: falls back to uint4 copies if the
// toolchain doesn't declare the builtins).
#if defined(__AMDGCN__) && __has_builtin(__builtin_amdgcn_global_load_async_to_lds_b128) && __has_builtin(__builtin_amdgcn_s_wait_asynccnt)
#define USE_ASYNC_LDS 1
#else
#define USE_ASYNC_LDS 0
#endif

#if defined(__AMDGCN__)
#define AS1 __attribute__((address_space(1)))
#define AS3 __attribute__((address_space(3)))
#endif

typedef __attribute__((ext_vector_type(16))) __bf16 v16bf;
typedef __attribute__((ext_vector_type(8)))  float  v8f;
typedef __attribute__((ext_vector_type(4)))  int    v4i;
union FragU { uint4 u[2]; v16bf v; };

// ---------------------------------------------------------------
// Kernel 1: row-wise L2 normalize fp32 -> bf16 (one wave per row)
// ---------------------------------------------------------------
__global__ void __launch_bounds__(256)
l2norm_bf16_kernel(const float* __restrict__ src,
                   __hip_bfloat16* __restrict__ dst, int rows) {
  const int lane = threadIdx.x & 31;
  const int wave = threadIdx.x >> 5;
  const int row  = blockIdx.x * 8 + wave;
  if (row >= rows) return;
  const float* s = src + (size_t)row * EDIM;
  float ss = 0.0f;
#pragma unroll
  for (int i = 0; i < EDIM / 32; ++i) {
    float v = s[lane + 32 * i];
    ss += v * v;
  }
#pragma unroll
  for (int off = 16; off > 0; off >>= 1) ss += __shfl_xor(ss, off, 32);
  const float inv = 1.0f / fmaxf(sqrtf(ss), 1e-12f);
  __hip_bfloat16* d = dst + (size_t)row * EDIM;
#pragma unroll
  for (int i = 0; i < EDIM / 32; ++i) {
    int e = lane + 32 * i;
    d[e] = __float2bfloat16(s[e] * inv);
  }
}

// ---------------------------------------------------------------
// Kernel 2: fused  bf16-WMMA sim GEMM + masked top-8 + softmax +
//           weighted gather of raw memory + residual add.
// ---------------------------------------------------------------
__global__ void __launch_bounds__(NTHR)
topk_attend_kernel(const __hip_bfloat16* __restrict__ qn,   // [NQ,E] normalized bf16
                   const int*            __restrict__ qlab, // [NQ]
                   const __hip_bfloat16* __restrict__ mn,   // [M,E] normalized bf16
                   const int*            __restrict__ mlabg,// [M]
                   const float*          __restrict__ memraw,// [M,E] fp32
                   const float*          __restrict__ x,    // [NQ,E] fp32
                   float*                __restrict__ out,  // [NQ,E] fp32
                   int M)
{
  extern __shared__ char smem[];
  __hip_bfloat16* qs  = (__hip_bfloat16*)(smem + QS_OFF);
  __hip_bfloat16* ms  = (__hip_bfloat16*)(smem + MS_OFF);
  float*          sim = (float*)(smem + SIM_OFF);
  int*            mlab = (int*)(smem + MLAB_OFF);
  float*          mgV = (float*)(smem + MGV_OFF);
  int*            mgI = (int*)(smem + MGI_OFF);
  float*          wv  = (float*)(smem + WV_OFF);
  int*            wi  = (int*)(smem + WI_OFF);

  const int t     = threadIdx.x;
  const int lane  = t & 31;
  const int wave  = t >> 5;
  const int qbase = blockIdx.x * QT;

  // ---- stage normalized query tile into LDS (32 rows x 768 bf16) ----
  {
    const int r = t >> 2, quarter = t & 3;   // 4 threads/row, 24 uint4 each
    const uint4* src = (const uint4*)(qn + (size_t)(qbase + r) * EDIM) + quarter * 24;
    uint4* dst = (uint4*)(qs + r * ROWP) + quarter * 24;
#pragma unroll
    for (int i = 0; i < 24; ++i) dst[i] = src[i];
  }

  // ---- per-thread running top-8 state (4 threads per query row) ----
  const int scanRow = t >> 2;                 // 0..31
  const int scanGrp = t & 3;                  // cols [grp*16, grp*16+16)
  const int myLab   = qlab[qbase + scanRow];
  float tv[TOPK]; int ti[TOPK];
#pragma unroll
  for (int j = 0; j < TOPK; ++j) { tv[j] = NEGINF; ti[j] = 0; }
  float rmin = NEGINF;

  // ---- per-lane WMMA fragment base pointers (A 16x32 / B 32x16 layouts) ----
  const int arow = lane & 15;
  const int sel  = lane >> 4;
  const __hip_bfloat16* a0p = qs + arow * ROWP;           // rows 0..15
  const __hip_bfloat16* a1p = qs + (16 + arow) * ROWP;    // rows 16..31
  const __hip_bfloat16* bp  = ms + (wave * 16 + arow) * ROWP;
  const int simcol = wave * 16 + arow;

  // chunk-staging per-thread addresses (2 threads/row, 48 x b128 each)
  const int stR = t >> 1, stH = t & 1;

  for (int base = 0; base < M; base += NCH) {
    __syncthreads();  // previous chunk's scan finished before overwriting ms

    // ---- stage 64 memory rows (bf16) + labels into LDS ----
    {
#if USE_ASYNC_LDS
      AS1 v4i* gsrc = (AS1 v4i*)(mn + (size_t)(base + stR) * EDIM + stH * 384);
      AS3 v4i* ldst = (AS3 v4i*)(ms + stR * ROWP + stH * 384);
#pragma unroll 8
      for (int i = 0; i < 48; ++i)
        __builtin_amdgcn_global_load_async_to_lds_b128(gsrc + i, ldst + i, 0, 0);
#else
      const uint4* src = (const uint4*)(mn + (size_t)(base + stR) * EDIM) + stH * 48;
      uint4* dst = (uint4*)(ms + stR * ROWP) + stH * 48;
#pragma unroll 4
      for (int i = 0; i < 48; ++i) dst[i] = src[i];
#endif
      if (t < NCH) mlab[t] = mlabg[base + t];
      if (base + NCH < M)
        __builtin_prefetch((const char*)(mn + (size_t)(base + NCH + stR) * EDIM), 0, 0);
#if USE_ASYNC_LDS
      __builtin_amdgcn_s_wait_asynccnt(0);
#endif
    }
    __syncthreads();

    // ---- 32x64 sim tile via bf16 WMMA, K = 768, double-buffered frags ----
    v8f acc0 = {0.f,0.f,0.f,0.f,0.f,0.f,0.f,0.f};
    v8f acc1 = {0.f,0.f,0.f,0.f,0.f,0.f,0.f,0.f};
    FragU fb[2], fa0[2], fa1[2];
#define LOADFRAGS(buf, k0)                                                  \
    do {                                                                    \
      fb[buf].u[0]  = *(const uint4*)(bp  + (k0) + sel * 8);                \
      fb[buf].u[1]  = *(const uint4*)(bp  + (k0) + 16 + sel * 8);           \
      fa0[buf].u[0] = *(const uint4*)(a0p + (k0) + sel * 8);                \
      fa0[buf].u[1] = *(const uint4*)(a0p + (k0) + 16 + sel * 8);           \
      fa1[buf].u[0] = *(const uint4*)(a1p + (k0) + sel * 8);                \
      fa1[buf].u[1] = *(const uint4*)(a1p + (k0) + 16 + sel * 8);           \
    } while (0)

    LOADFRAGS(0, 0);
#pragma unroll
    for (int kk = 0; kk < KSTEPS; ++kk) {
      const int cur = kk & 1;
      const int nxt = cur ^ 1;
      if (kk + 1 < KSTEPS) LOADFRAGS(nxt, (kk + 1) * 32);
      acc0 = __builtin_amdgcn_wmma_f32_16x16x32_bf16(
                 false, fa0[cur].v, false, fb[cur].v, (short)0, acc0, false, false);
      acc1 = __builtin_amdgcn_wmma_f32_16x16x32_bf16(
                 false, fa1[cur].v, false, fb[cur].v, (short)0, acc1, false, false);
    }
#undef LOADFRAGS

    // C/D layout: VGPR r -> row (r + 8*sel), col = lane&15 within wave tile
#pragma unroll
    for (int r = 0; r < 8; ++r) {
      sim[(r + 8 * sel) * SIMS + simcol]      = acc0[r];
      sim[(16 + r + 8 * sel) * SIMS + simcol] = acc1[r];
    }
    __syncthreads();

    // ---- label-masked running top-8 update (16 candidates / thread) ----
#pragma unroll 4
    for (int j = 0; j < 16; ++j) {
      const int   c  = scanGrp * 16 + j;
      const float v  = sim[scanRow * SIMS + c];
      const bool  ok = (mlab[c] == myLab);
      if (ok && v > rmin) {
        float mn2 = tv[0]; int mp = 0;
#pragma unroll
        for (int s2 = 1; s2 < TOPK; ++s2) {
          bool lt = tv[s2] < mn2;
          mn2 = lt ? tv[s2] : mn2;
          mp  = lt ? s2 : mp;
        }
#pragma unroll
        for (int s2 = 0; s2 < TOPK; ++s2)
          if (s2 == mp) { tv[s2] = v; ti[s2] = base + c; }
        rmin = tv[0];
#pragma unroll
        for (int s2 = 1; s2 < TOPK; ++s2) rmin = fminf(rmin, tv[s2]);
      }
    }
  }

  __syncthreads();  // ms region dead; safe to alias merge buffers
#pragma unroll
  for (int j = 0; j < TOPK; ++j) {
    mgV[scanRow * 32 + scanGrp * 8 + j] = tv[j];
    mgI[scanRow * 32 + scanGrp * 8 + j] = ti[j];
  }
  __syncthreads();

  // ---- merge 4x8 -> top-8 per row, then softmax weights ----
  if (t < QT) {
    float bv[TOPK]; int bi[TOPK];
#pragma unroll
    for (int k = 0; k < TOPK; ++k) {
      float best = -3.0e38f; int bpj = 0;
      for (int j = 0; j < 32; ++j) {
        float v = mgV[t * 32 + j];
        if (v > best) { best = v; bpj = j; }
      }
      bv[k] = best; bi[k] = mgI[t * 32 + bpj];
      mgV[t * 32 + bpj] = -3.0e38f;
    }
    const float mx = bv[0];
    float ex[TOPK]; float s = 0.f;
#pragma unroll
    for (int k = 0; k < TOPK; ++k) { ex[k] = __expf(bv[k] - mx); s += ex[k]; }
    const float invs = 1.0f / s;
#pragma unroll
    for (int k = 0; k < TOPK; ++k) { wv[t * TOPK + k] = ex[k] * invs; wi[t * TOPK + k] = bi[k]; }
  }
  __syncthreads();

  // ---- weighted gather of raw memory rows + residual (coalesced) ----
  for (int r = 0; r < QT; ++r) {
    float wk[TOPK]; int ik[TOPK];
#pragma unroll
    for (int k = 0; k < TOPK; ++k) { wk[k] = wv[r * TOPK + k]; ik[k] = wi[r * TOPK + k]; }
    const size_t qoff = (size_t)(qbase + r) * EDIM;
    for (int e = t; e < EDIM; e += NTHR) {
      float acc = x[qoff + e];
#pragma unroll
      for (int k = 0; k < TOPK; ++k)
        acc += wk[k] * memraw[(size_t)ik[k] * EDIM + e];
      out[qoff + e] = acc;
    }
  }
}

// ---------------------------------------------------------------
extern "C" void kernel_launch(void* const* d_in, const int* in_sizes, int n_in,
                              void* d_out, int out_size, void* d_ws, size_t ws_size,
                              hipStream_t stream) {
  (void)n_in; (void)out_size; (void)ws_size;
  const float* x       = (const float*)d_in[0];
  const int*   labels  = (const int*)d_in[1];
  const float* memory  = (const float*)d_in[2];
  const int*   mlabels = (const int*)d_in[3];
  float*       out     = (float*)d_out;

  const int NQ = in_sizes[0] / EDIM;   // B*S = 8192
  const int M  = in_sizes[2] / EDIM;   // 32768

  __hip_bfloat16* mn = (__hip_bfloat16*)d_ws;
  __hip_bfloat16* qn = (__hip_bfloat16*)((char*)d_ws + (size_t)M * EDIM * 2);

  l2norm_bf16_kernel<<<(M + 7) / 8, 256, 0, stream>>>(memory, mn, M);
  l2norm_bf16_kernel<<<(NQ + 7) / 8, 256, 0, stream>>>(x, qn, NQ);

  topk_attend_kernel<<<NQ / QT, NTHR, SMEM_TOTAL, stream>>>(
      qn, labels, mn, mlabels, memory, x, out, M);
}
